// CausalSelfAttention_30829275251122
// MI455X (gfx1250) — matrix operand order repaired and checked by reading, not compiled, not run on hardware
//
#include <hip/hip_runtime.h>

// ---------------------------------------------------------------------------
// Causal self-attention for gfx1250 (MI455X): f16 WMMA pipeline, f32 softmax.
//   B=4, T=2048, C=1024, H=16, hd=64
// CDNA5 paths: v_wmma_f32_16x16x32_f16, tensor_load_to_lds (TDM, double-
// buffered GEMM B-slab), global_load_async_to_lds_b128 (pipelined attention
// K tiles), ds_swizzle butterfly reductions, s_wait_{async,tensor,ds}cnt.
// ---------------------------------------------------------------------------

typedef __attribute__((ext_vector_type(16))) _Float16 v16h;
typedef __attribute__((ext_vector_type(8)))  _Float16 v8h;
typedef __attribute__((ext_vector_type(8)))  float    v8f;
typedef __attribute__((ext_vector_type(4)))  unsigned int v4u;
typedef __attribute__((ext_vector_type(8)))  int v8i;
typedef __attribute__((ext_vector_type(4)))  int v4i;

#define N_EMBD   1024
#define N_HEADS  16
#define HEAD_DIM 64
#define BATCH    4
#define SEQ      2048

// ------------------------- fragment helpers --------------------------------

__device__ __forceinline__ v16h concat8(v8h lo, v8h hi) {
  return __builtin_shufflevector(lo, hi, 0,1,2,3,4,5,6,7,8,9,10,11,12,13,14,15);
}

// A-matrix fragment (16x32 f16): lanes 0-15 K=[0..7],[16..23]; lanes 16-31
// K=[8..15],[24..31]. `row` points at this lane's matrix-row start.
__device__ __forceinline__ v16h load_a_frag(const _Float16* row, int lane) {
  const int koff = (lane >> 4) << 3;
  v8h lo = *(const v8h*)(row + koff);
  v8h hi = *(const v8h*)(row + 16 + koff);
  return concat8(lo, hi);
}

// B-matrix fragment (32x16 f16) from transposed storage: lanes 0-15 hold
// K=[0..7],[16..23]; lanes 16-31 K=[8..15],[24..31] (same order as A).
__device__ __forceinline__ v16h load_b_frag(const _Float16* row, int lane) {
  const int koff = (lane >> 4) << 3;
  v8h lo = *(const v8h*)(row + koff);
  v8h hi = *(const v8h*)(row + 16 + koff);
  return concat8(lo, hi);
}

// ------------------------- async / TDM helpers -----------------------------

// 16B global -> LDS async copy (ASYNCcnt-tracked).
__device__ __forceinline__ void async_copy_b128(const void* gptr, void* lptr) {
  unsigned int  l = (unsigned int)(unsigned long long)(uintptr_t)lptr;  // LDS offset
  unsigned long long g = (unsigned long long)(uintptr_t)gptr;
  asm volatile("global_load_async_to_lds_b128 %0, %1, off" :: "v"(l), "v"(g) : "memory");
}
#define ASYNC_WAIT(n) asm volatile("s_wait_asynccnt " #n ::: "memory")

#if __has_builtin(__builtin_amdgcn_tensor_load_to_lds)
#define HAVE_TDM 1
// 2D tile DMA: tile_d0 x tile_d1 (elements, 2B each), row pitch stride_elems.
// Descriptor per CDNA5 ISA 8.3/8.4 (D# group0 + group1; groups 2/3 zero).
// This toolchain's builtin takes 6 args: (v4u, v8i, v4i, v4i, v8i, i32 cpol).
__device__ __forceinline__ void tdm_load_2d(const void* gptr, void* lptr,
                                            unsigned tensor_d0, unsigned tensor_d1,
                                            unsigned tile_d0, unsigned tile_d1,
                                            unsigned stride_elems) {
  unsigned long long g = (unsigned long long)(uintptr_t)gptr;
  unsigned int l = (unsigned int)(unsigned long long)(uintptr_t)lptr;
  v4u g0 = { 1u,                                   // count=1, user mode
             l,                                    // lds_addr
             (unsigned)g,                          // global_addr[31:0]
             (unsigned)((g >> 32) & 0x01FFFFFFu) | (2u << 30) };  // addr[56:32]|type=2
  v8i g1 = { (int)0x00010000u,                                   // data_size=2B
             (int)((tensor_d0 & 0xFFFFu) << 16),                 // tensor_dim0 lo
             (int)(((tensor_d0 >> 16) & 0xFFFFu) | ((tensor_d1 & 0xFFFFu) << 16)),
             (int)(((tensor_d1 >> 16) & 0xFFFFu) | ((tile_d0 & 0xFFFFu) << 16)),
             (int)(tile_d1 & 0xFFFFu),                           // tile_dim1 (tile_dim2=0)
             (int)stride_elems,                                  // tensor_dim0_stride lo32
             0, 0 };
  v4i z4 = { 0, 0, 0, 0 };
  v8i z8 = { 0, 0, 0, 0, 0, 0, 0, 0 };
  __builtin_amdgcn_tensor_load_to_lds(g0, g1, z4, z4, z8, 0);
}
#endif

// ------------------------- swizzle reductions ------------------------------
// Butterfly over the 16-lane halves of a wave32; 8 rows batched per stage so
// ds ops clause together under one s_wait_dscnt.

#define SWZF(x, imm) \
  __int_as_float(__builtin_amdgcn_ds_swizzle(__float_as_int(x), imm))

#define RED16_STAGE(v, imm, OP) do {                                          \
    float t0_=SWZF(v[0],imm), t1_=SWZF(v[1],imm), t2_=SWZF(v[2],imm),          \
          t3_=SWZF(v[3],imm), t4_=SWZF(v[4],imm), t5_=SWZF(v[5],imm),          \
          t6_=SWZF(v[6],imm), t7_=SWZF(v[7],imm);                              \
    v[0]=OP(v[0],t0_); v[1]=OP(v[1],t1_); v[2]=OP(v[2],t2_); v[3]=OP(v[3],t3_);\
    v[4]=OP(v[4],t4_); v[5]=OP(v[5],t5_); v[6]=OP(v[6],t6_); v[7]=OP(v[7],t7_);\
  } while (0)

#define ADDF(a, b) ((a) + (b))
#define RED16(v, OP) do {                \
    RED16_STAGE(v, 0x041F, OP);          \
    RED16_STAGE(v, 0x081F, OP);          \
    RED16_STAGE(v, 0x101F, OP);          \
    RED16_STAGE(v, 0x201F, OP);          \
  } while (0)

// ------------------------------- converts ----------------------------------

__global__ void convert_f32_to_f16(const float* __restrict__ in,
                                   _Float16* __restrict__ out, size_t n) {
  size_t i = blockIdx.x * 256ull + threadIdx.x;
  if (i < n) out[i] = (_Float16)in[i];
}

// in: [K, N] f32 row-major -> out: [N, K] f16 row-major (transposed)
__global__ void transpose_to_f16(const float* __restrict__ in,
                                 _Float16* __restrict__ out, int K, int N) {
  size_t idx = blockIdx.x * 256ull + threadIdx.x;
  if (idx >= (size_t)K * N) return;
  int k = (int)(idx / N);
  int n = (int)(idx % N);
  out[(size_t)n * K + k] = (_Float16)in[idx];
}

// ------------------------------- WMMA GEMM ---------------------------------
// C[M,N] = A[M,K](f16 row-major) * Bt[N,K](f16 row-major, = B transposed).
// 256 threads = 8 waves -> 128(M) x 64(N) tile; B k-slab (64x64) staged in
// LDS by TDM (wave 0), double-buffered with s_wait_tensorcnt.

#define GK 64

template <bool OUT_F32>
__global__ void gemm_wmma_f16(const _Float16* __restrict__ A,
                              const _Float16* __restrict__ Bt,
                              void* __restrict__ Cout,
                              int M, int N, int K) {
  const int lane = threadIdx.x & 31;
  const int wave = threadIdx.x >> 5;
  const int rowBase = blockIdx.y * 128 + wave * 16;
  const int colBase = blockIdx.x * 64;
  const int r = lane & 15;

  __shared__ __align__(16) _Float16 Bs[2][64][GK];   // 16 KB

  auto stageB = [&](int buf, int k0) {
#ifdef HAVE_TDM
    if (wave == 0)
      tdm_load_2d(Bt + (size_t)colBase * K + k0, &Bs[buf][0][0],
                  (unsigned)K, (unsigned)N, GK, 64, (unsigned)K);
#else
    const int col = threadIdx.x >> 2;          // 0..63
    const int kk  = (threadIdx.x & 3) << 4;    // 0,16,32,48
    const _Float16* g = Bt + (size_t)(colBase + col) * K + k0 + kk;
    async_copy_b128(g,     &Bs[buf][col][kk]);
    async_copy_b128(g + 8, &Bs[buf][col][kk + 8]);
#endif
  };

  v8f acc[4] = {};
  const _Float16* arow = A + (size_t)(rowBase + r) * K;

  stageB(0, 0);
  for (int k0 = 0; k0 < K; k0 += GK) {
    const int buf = (k0 / GK) & 1;
    if (k0 + GK < K) {
      stageB(buf ^ 1, k0 + GK);
#ifdef HAVE_TDM
      __builtin_amdgcn_s_wait_tensorcnt(1);
#else
      ASYNC_WAIT(2);
#endif
    } else {
#ifdef HAVE_TDM
      __builtin_amdgcn_s_wait_tensorcnt(0);
#else
      ASYNC_WAIT(0);
#endif
    }
    __syncthreads();

#pragma unroll
    for (int kk = 0; kk < GK; kk += 32) {
      if (k0 + kk + 64 < K) __builtin_prefetch(arow + k0 + kk + 64, 0, 0);
      v16h a = load_a_frag(arow + k0 + kk, lane);
#pragma unroll
      for (int n = 0; n < 4; ++n) {
        v16h b = load_b_frag(&Bs[buf][n * 16 + r][kk], lane);
        acc[n] = __builtin_amdgcn_wmma_f32_16x16x32_f16(
            false, a, false, b, (short)0, acc[n], false, false);
      }
    }
    __syncthreads();   // protect buf from being re-staged while still read
  }

  const int orow = rowBase + ((lane >> 4) << 3);
  if (OUT_F32) {
    float* C = (float*)Cout;
#pragma unroll
    for (int n = 0; n < 4; ++n)
#pragma unroll
      for (int rr = 0; rr < 8; ++rr)
        C[(size_t)(orow + rr) * N + colBase + n * 16 + r] = acc[n][rr];
  } else {
    _Float16* C = (_Float16*)Cout;
#pragma unroll
    for (int n = 0; n < 4; ++n)
#pragma unroll
      for (int rr = 0; rr < 8; ++rr)
        C[(size_t)(orow + rr) * N + colBase + n * 16 + r] = (_Float16)acc[n][rr];
  }
}

// --------------------------- flash attention -------------------------------
// qkv f16 [B*T, 3C]; per (b,h): Q/K/V [T,64] strided views.
// 128 query rows per WG (8 waves x 16), key tiles of 32, 2-deep software
// pipeline: async K(i+1) + V(i+1) global load issued before compute(i);
// V(i+1) transposed-stored to LDS after compute(i).

__global__ void attn_flash_wmma(const _Float16* __restrict__ qkv,
                                _Float16* __restrict__ out) {
  const int lane = threadIdx.x & 31;
  const int wave = threadIdx.x >> 5;
  const int r = lane & 15;
  const int qhi = (lane >> 4) << 3;

  const int qtPerSeq = SEQ / 128;
  int blk = blockIdx.x;
  const int qt = blk % qtPerSeq;  blk /= qtPerSeq;
  const int h  = blk % N_HEADS;   blk /= N_HEADS;
  const int b  = blk;

  const size_t rowStride = 3 * N_EMBD;
  const _Float16* qbase = qkv + (size_t)b * SEQ * rowStride + h * HEAD_DIM;
  const _Float16* kbase = qbase + N_EMBD;
  const _Float16* vbase = qbase + 2 * N_EMBD;

  __shared__ __align__(16) _Float16 Ks[2][32][HEAD_DIM];  // K tiles       8 KB
  __shared__ __align__(16) _Float16 Vt[2][HEAD_DIM][32];  // V^T tiles     8 KB
  __shared__ __align__(16) _Float16 P[8][16][32];         // per-wave P    8 KB

  const int q0 = qt * 128 + wave * 16;
  const int q0s = __builtin_amdgcn_readfirstlane(q0);     // scalar for branch

  const _Float16* qrow = qbase + (size_t)(q0 + r) * rowStride;
  const v16h aq0 = load_a_frag(qrow,      lane);          // d = 0..31
  const v16h aq1 = load_a_frag(qrow + 32, lane);          // d = 32..63

  // staging roles: thread -> (key row, dim chunk)
  const int skk = threadIdx.x >> 3;                       // 0..31
  const int sd0 = (threadIdx.x & 7) * 8;                  // 0,8,...,56

  auto asyncK = [&](int buf, int kt2) {
    async_copy_b128(kbase + (size_t)(kt2 + skk) * rowStride + sd0,
                    &Ks[buf][skk][sd0]);
  };
  auto loadV = [&](int kt2) -> v8h {
    return *(const v8h*)(vbase + (size_t)(kt2 + skk) * rowStride + sd0);
  };
  auto storeVt = [&](int buf, v8h vv) {
#pragma unroll
    for (int i = 0; i < 8; ++i) Vt[buf][sd0 + i][skk] = vv[i];
  };

  float m[8], l[8];
  v8f o[4] = {};
#pragma unroll
  for (int i = 0; i < 8; ++i) { m[i] = -1e30f; l[i] = 0.0f; }

  // pipeline prologue: stage tile 0 into buffer 0
  asyncK(0, 0);
  storeVt(0, loadV(0));

  const int kend = qt * 128 + 128;
  for (int kt = 0; kt < kend; kt += 32) {
    const int buf = (kt >> 5) & 1;
    const bool hasNext = (kt + 32) < kend;

    v8h vnext = {};
    if (hasNext) {
      asyncK(buf ^ 1, kt + 32);          // async DMA: K(i+1) -> LDS
      vnext = loadV(kt + 32);            // V(i+1) -> VGPRs (latency hidden)
      ASYNC_WAIT(1);                     // K(i) complete; K(i+1) in flight
    } else {
      ASYNC_WAIT(0);
    }
    __syncthreads();

    if (kt <= q0s + 15) {                // scalar branch: EXEC stays ~0
      // ---- S = Q K^T (16q x 32k) ----
      v8f s[2] = {};
#pragma unroll
      for (int j = 0; j < 2; ++j) {
        v16h b0 = load_b_frag(&Ks[buf][j * 16 + r][0],  lane);  // d = 0..31
        v16h b1 = load_b_frag(&Ks[buf][j * 16 + r][32], lane);  // d = 32..63
        s[j] = __builtin_amdgcn_wmma_f32_16x16x32_f16(
            false, aq0, false, b0, (short)0, s[j], false, false);
        s[j] = __builtin_amdgcn_wmma_f32_16x16x32_f16(
            false, aq1, false, b1, (short)0, s[j], false, false);
      }

      // ---- causal mask + online softmax ----
      const float sc = 0.125f;                          // 1/sqrt(64)
      float v0a[8], v1a[8], mx[8];
#pragma unroll
      for (int rr = 0; rr < 8; ++rr) {
        const int q = q0 + rr + qhi;
        float a0 = s[0][rr] * sc; if (kt + r      > q) a0 = -1e30f;
        float a1 = s[1][rr] * sc; if (kt + 16 + r > q) a1 = -1e30f;
        v0a[rr] = a0; v1a[rr] = a1; mx[rr] = fmaxf(a0, a1);
      }
      RED16(mx, fmaxf);

      float rs[8], corr[8];
#pragma unroll
      for (int rr = 0; rr < 8; ++rr) {
        const float mnew = fmaxf(m[rr], mx[rr]);
        const float p0 = __expf(v0a[rr] - mnew);
        const float p1 = __expf(v1a[rr] - mnew);
        corr[rr] = __expf(m[rr] - mnew);
        m[rr] = mnew;
        rs[rr] = p0 + p1;
        P[wave][rr + qhi][r]      = (_Float16)p0;       // C-layout -> LDS
        P[wave][rr + qhi][16 + r] = (_Float16)p1;
      }
      RED16(rs, ADDF);
#pragma unroll
      for (int rr = 0; rr < 8; ++rr) {
        l[rr] = l[rr] * corr[rr] + rs[rr];
#pragma unroll
        for (int n = 0; n < 4; ++n) o[n][rr] = o[n][rr] * corr[rr];
      }
      asm volatile("s_wait_dscnt 0" ::: "memory");      // P visible for reload

      // ---- O += P V ----
      v16h ap = load_a_frag(&P[wave][r][0], lane);
#pragma unroll
      for (int n = 0; n < 4; ++n) {
        v16h bv = load_b_frag(&Vt[buf][n * 16 + r][0], lane);
        o[n] = __builtin_amdgcn_wmma_f32_16x16x32_f16(
            false, ap, false, bv, (short)0, o[n], false, false);
      }
    }

    if (hasNext) storeVt(buf ^ 1, vnext);  // transpose V(i+1) into other buffer
    __syncthreads();
  }

  // ---- epilogue ----
#pragma unroll
  for (int rr = 0; rr < 8; ++rr) {
    const int q = q0 + rr + qhi;
    const float inv = 1.0f / l[rr];
#pragma unroll
    for (int n = 0; n < 4; ++n) {
      out[(size_t)(b * SEQ + q) * N_EMBD + h * HEAD_DIM + n * 16 + r] =
          (_Float16)(o[n][rr] * inv);
    }
  }
}

// ------------------------------- launch ------------------------------------

extern "C" void kernel_launch(void* const* d_in, const int* in_sizes, int n_in,
                              void* d_out, int out_size, void* d_ws, size_t ws_size,
                              hipStream_t stream) {
  const float* x      = (const float*)d_in[0];   // [4,2048,1024]
  const float* W_attn = (const float*)d_in[1];   // [1024,3072]
  const float* W_proj = (const float*)d_in[2];   // [1024,1024]
  float* out = (float*)d_out;                    // [4,2048,1024]

  const size_t M  = (size_t)BATCH * SEQ;         // 8192
  const size_t C  = N_EMBD;                      // 1024
  const size_t C3 = 3 * N_EMBD;                  // 3072

  // f16 workspace: x | W_attn^T | W_proj^T | qkv | attn_out  (~88 MB)
  _Float16* xh   = (_Float16*)d_ws;
  _Float16* waT  = xh   + M * C;
  _Float16* wpT  = waT  + C3 * C;
  _Float16* qkvh = wpT  + C * C;
  _Float16* atth = qkvh + M * C3;

  convert_f32_to_f16<<<dim3((unsigned)((M * C) / 256)), 256, 0, stream>>>(x, xh, M * C);
  transpose_to_f16<<<dim3((unsigned)((C * C3) / 256)), 256, 0, stream>>>(W_attn, waT, (int)C, (int)C3);
  transpose_to_f16<<<dim3((unsigned)((C * C) / 256)), 256, 0, stream>>>(W_proj, wpT, (int)C, (int)C);

  gemm_wmma_f16<false><<<dim3((unsigned)(C3 / 64), (unsigned)(M / 128)), 256, 0, stream>>>(
      xh, waT, qkvh, (int)M, (int)C3, (int)C);

  attn_flash_wmma<<<dim3(BATCH * N_HEADS * (SEQ / 128)), 256, 0, stream>>>(qkvh, atth);

  gemm_wmma_f16<true><<<dim3((unsigned)(C / 64), (unsigned)(M / 128)), 256, 0, stream>>>(
      atth, wpT, out, (int)M, (int)C, (int)C);
}
